// TransducerOut_3925600108640
// MI455X (gfx1250) — compile-verified, matrix operand order checked by reading, not run on hardware
//
#include <hip/hip_runtime.h>

// Problem constants (from reference)
#define BB 8
#define TT 128
#define UU1 33
#define UU (UU1 - 1)
#define DD 512
#define VV 4096
#define RTOT (BB * TT * UU1)      // 33792 rows of x
#define PADV 1
#define LSMOOTH 0.1f

// GEMM tiling
#define ROWS_PB 48                 // rows per block (3 row-tiles of 16)
#define NTHREADS 96                // 3 waves (wave32)
#define KSTEPS (DD / 32)           // 16 k-steps of K=32
#define NTILES (VV / 16)           // 256 column tiles
#define LDA 520                    // padded LDS stride in halves (1040B -> bank-friendly)
#define LDB 520
#define CE_ROWS 256
#define CE_ROWS_PAD 288            // 6 blocks * 48

// Async global->LDS path (CDNA5): guarded so both compile passes/toolchains stay green.
#if defined(__HIP_DEVICE_COMPILE__) && defined(__has_builtin)
#if __has_builtin(__builtin_amdgcn_global_load_async_to_lds_b128) && \
    __has_builtin(__builtin_amdgcn_s_wait_asynccnt)
#define USE_ASYNC_LDS 1
#endif
#endif
#ifndef USE_ASYNC_LDS
#define USE_ASYNC_LDS 0
#endif

#define AS1 __attribute__((address_space(1)))
#define AS3 __attribute__((address_space(3)))

typedef __attribute__((__vector_size__(4 * sizeof(int)))) int v4i;

typedef __attribute__((ext_vector_type(16))) __bf16 v16bf;
typedef __attribute__((ext_vector_type(8)))  __bf16 v8bf;
typedef __attribute__((ext_vector_type(8)))  float  v8f;
typedef __attribute__((ext_vector_type(8)))  unsigned short v8u;

static __device__ __forceinline__ v16bf cat16(v8bf a, v8bf b) {
  return __builtin_shufflevector(a, b, 0, 1, 2, 3, 4, 5, 6, 7, 8, 9, 10, 11, 12, 13, 14, 15);
}

// ---------------- prep: split weight into bf16 hi + bf16 residual ----------------
__global__ __launch_bounds__(256) void k_prep_weight(const float* __restrict__ w,
                                                     unsigned short* __restrict__ whi,
                                                     unsigned short* __restrict__ wlo) {
  int i = (blockIdx.x * 256 + threadIdx.x) * 8;
  __bf16* Whi = (__bf16*)whi;
  __bf16* Wlo = (__bf16*)wlo;
#pragma unroll
  for (int j = 0; j < 8; ++j) {
    float f = w[i + j];
    __bf16 h = (__bf16)f;
    Whi[i + j] = h;
    Wlo[i + j] = (__bf16)(f - (float)h);
  }
}

// ---------------- prep: target-column per row, CE row gather map ----------------
__global__ __launch_bounds__(256) void k_prep_index(const int* __restrict__ targets,
                                                    const int* __restrict__ srcl,
                                                    int* __restrict__ tcolR,
                                                    int* __restrict__ rowmap,
                                                    int* __restrict__ tcolCE) {
  int r = blockIdx.x * 256 + threadIdx.x;
  if (r < RTOT) {
    int b = r / (TT * UU1);
    int u = r % UU1;
    tcolR[r] = (u < UU) ? targets[b * UU + u] : -1;
  }
  if (r < CE_ROWS_PAD) {
    if (r < CE_ROWS) {
      int b = r / UU, u = r % UU;
      rowmap[r] = (b * TT + (srcl[b] - 1)) * UU1 + u;   // flattened x row index
      tcolCE[r] = targets[r];                            // r == b*UU+u
    } else {
      rowmap[r] = 0;
      tcolCE[r] = -1;
    }
  }
}

// ---------------- fused GEMM + online softmax stats ----------------
// mode 0: RNNT rows (direct), write blank_lp/emit_lp
// mode 1: CE rows (gathered), write per-row smoothed-CE contribution
__global__ __launch_bounds__(NTHREADS) void k_gemm(const float* __restrict__ x,
                                                   const unsigned short* __restrict__ whi,
                                                   const unsigned short* __restrict__ wlo,
                                                   const float* __restrict__ bias,
                                                   const int* __restrict__ tcArr,
                                                   const int* __restrict__ rowmap,
                                                   float* __restrict__ blank_lp,
                                                   float* __restrict__ emit_lp,
                                                   float* __restrict__ ce_contrib,
                                                   int mode) {
  __shared__ __align__(16) unsigned short sAh[ROWS_PB * LDA];
  __shared__ __align__(16) unsigned short sAl[ROWS_PB * LDA];
  __shared__ __align__(16) unsigned short sBh[16 * LDB];
  __shared__ __align__(16) unsigned short sBl[16 * LDB];

  const int tid = threadIdx.x;
  const int wave = tid >> 5;
  const int lane = tid & 31;
  const int hf = lane >> 4;        // lane half: C rows e / e+8, K-block select
  const int l16 = lane & 15;
  const int row0 = blockIdx.x * ROWS_PB;

  // Stage A: this block's 48 rows of x, split to bf16 hi/lo in LDS.
  for (int idx = tid; idx < ROWS_PB * (DD / 8); idx += NTHREADS) {
    int rloc = idx >> 6;           // /64 chunks of 8
    int ch = idx & 63;
    int rglob = row0 + rloc;
    long rrow = mode ? (long)rowmap[rglob] : (long)rglob;
    const float* src = x + rrow * (long)DD + ch * 8;
    __bf16* ah = (__bf16*)&sAh[rloc * LDA + ch * 8];
    __bf16* al = (__bf16*)&sAl[rloc * LDA + ch * 8];
#pragma unroll
    for (int j = 0; j < 8; ++j) {
      float f = src[j];
      __bf16 h = (__bf16)f;
      ah[j] = h;
      al[j] = (__bf16)(f - (float)h);
    }
  }

  // Per-lane state for this wave's 8 rows (C layout: lanes 0-15 -> M=e, 16-31 -> M=e+8)
  float m[8], s[8], blk[8], tgt[8], ssum[8];
  int tc[8];
#pragma unroll
  for (int e = 0; e < 8; ++e) {
    m[e] = -3.4e38f;
    s[e] = 0.0f;
    blk[e] = 0.0f;
    tgt[e] = 0.0f;
    ssum[e] = 0.0f;
    tc[e] = tcArr[row0 + wave * 16 + hf * 8 + e];
  }
  __syncthreads();

  const int arow = wave * 16 + l16;   // A fragment: lane holds row l16 of this wave's tile
  const int kbase = hf * 8;           // ISA 16-bit A layout: halves hold K base..base+7, base+16..base+23

  for (int nt = 0; nt < NTILES; ++nt) {
    const int n0 = nt * 16;
    // Stage B tile (16 weight columns, hi+lo).
    for (int idx = tid; idx < 16 * (DD / 8); idx += NTHREADS) {
      int col = idx >> 6;
      int ch = idx & 63;
      size_t goff = (size_t)(n0 + col) * DD + ch * 8;
      int loff = col * LDB + ch * 8;
#if USE_ASYNC_LDS
      // CDNA5 async pipe: cache -> LDS directly, tracked by ASYNCcnt.
      __builtin_amdgcn_global_load_async_to_lds_b128(
          (AS1 v4i*)&whi[goff], (AS3 v4i*)&sBh[loff], 0, 0);
      __builtin_amdgcn_global_load_async_to_lds_b128(
          (AS1 v4i*)&wlo[goff], (AS3 v4i*)&sBl[loff], 0, 0);
#else
      v8u h = *(const v8u*)&whi[goff];
      v8u l = *(const v8u*)&wlo[goff];
      *(v8u*)&sBh[loff] = h;
      *(v8u*)&sBl[loff] = l;
#endif
    }
#if USE_ASYNC_LDS
    __builtin_amdgcn_s_wait_asynccnt(0);
#endif
    __syncthreads();
    if (tid == 0 && nt + 1 < NTILES) {
      __builtin_prefetch(&whi[(size_t)(n0 + 16) * DD], 0, 1);  // global_prefetch_b8
    }

    v8f acc = {0.f, 0.f, 0.f, 0.f, 0.f, 0.f, 0.f, 0.f};
#pragma unroll 4
    for (int ks = 0; ks < KSTEPS; ++ks) {
      const int ko = ks * 32 + kbase;
      v8bf ah0 = *(const v8bf*)&sAh[arow * LDA + ko];
      v8bf ah1 = *(const v8bf*)&sAh[arow * LDA + ko + 16];
      v8bf al0 = *(const v8bf*)&sAl[arow * LDA + ko];
      v8bf al1 = *(const v8bf*)&sAl[arow * LDA + ko + 16];
      v8bf bh0 = *(const v8bf*)&sBh[l16 * LDB + ko];
      v8bf bh1 = *(const v8bf*)&sBh[l16 * LDB + ko + 16];
      v8bf bl0 = *(const v8bf*)&sBl[l16 * LDB + ko];
      v8bf bl1 = *(const v8bf*)&sBl[l16 * LDB + ko + 16];
      v16bf ah = cat16(ah0, ah1), al = cat16(al0, al1);
      v16bf bh = cat16(bh0, bh1), bl = cat16(bl0, bl1);
      // 3xBF16 split: hi*hi + hi*lo + lo*hi  (~fp32 accuracy), f32 accumulate
      acc = __builtin_amdgcn_wmma_f32_16x16x32_bf16(false, ah, false, bh, (short)0, acc, false, false);
      acc = __builtin_amdgcn_wmma_f32_16x16x32_bf16(false, ah, false, bl, (short)0, acc, false, false);
      acc = __builtin_amdgcn_wmma_f32_16x16x32_bf16(false, al, false, bh, (short)0, acc, false, false);
    }

    const int ncol = n0 + l16;
    const float bn = bias[ncol];
#pragma unroll
    for (int e = 0; e < 8; ++e) {
      float v = acc[e] + bn;
      if (ncol == 0) blk[e] = v;
      if (ncol == tc[e]) tgt[e] = v;
      if (mode) ssum[e] += v;
      // online log-sum-exp (one expf per update)
      if (v <= m[e]) {
        s[e] += expf(v - m[e]);
      } else {
        s[e] = s[e] * expf(m[e] - v) + 1.0f;
        m[e] = v;
      }
    }
    __syncthreads();
  }

  // Butterfly merge across the 16 lanes that share each row (xor 1,2,4,8 stays in-half)
#pragma unroll
  for (int off = 1; off <= 8; off <<= 1) {
#pragma unroll
    for (int e = 0; e < 8; ++e) {
      float om = __shfl_xor(m[e], off, 32);
      float os = __shfl_xor(s[e], off, 32);
      float nm = fmaxf(m[e], om);
      s[e] = s[e] * expf(m[e] - nm) + os * expf(om - nm);
      m[e] = nm;
      blk[e] += __shfl_xor(blk[e], off, 32);
      tgt[e] += __shfl_xor(tgt[e], off, 32);
      if (mode) ssum[e] += __shfl_xor(ssum[e], off, 32);
    }
  }

  if (l16 == 0) {
#pragma unroll
    for (int e = 0; e < 8; ++e) {
      int r = row0 + wave * 16 + hf * 8 + e;
      float lse = m[e] + logf(s[e]);
      if (!mode) {
        blank_lp[r] = blk[e] - lse;
        if (tc[e] >= 0) emit_lp[r] = tgt[e] - lse;
      } else if (r < CE_ROWS) {
        float nll = lse - tgt[e];
        float smooth = (float)VV * lse - ssum[e];
        float eps_i = LSMOOTH / (float)(VV - 1);
        float msk = (tc[e] != PADV) ? 1.0f : 0.0f;
        ce_contrib[r] = ((1.0f - LSMOOTH - eps_i) * nll + eps_i * smooth) * msk;
      }
    }
  }
}

// ---------------- RNNT DP + final reduction ----------------
static __device__ __forceinline__ float lae(float a, float b) {
  float mx = fmaxf(a, b), mn = fminf(a, b);
  return mx + log1pf(expf(mn - mx));
}

__global__ __launch_bounds__(256) void k_dp(const float* __restrict__ blank_lp,
                                            const float* __restrict__ emit_lp,
                                            const float* __restrict__ ce_contrib,
                                            const int* __restrict__ srcl,
                                            const int* __restrict__ tgtl,
                                            float* __restrict__ out) {
  __shared__ float red[256];
  __shared__ float alpha[BB][UU1];
  __shared__ float nllb[BB];
  int tid = threadIdx.x;
  red[tid] = ce_contrib[tid];
  __syncthreads();
  for (int st = 128; st > 0; st >>= 1) {
    if (tid < st) red[tid] += red[tid + st];
    __syncthreads();
  }
  if (tid < BB) {
    int b = tid;
    int S = srcl[b], U = tgtl[b];
    const float* bl = blank_lp + (size_t)b * TT * UU1;
    const float* em = emit_lp + (size_t)b * TT * UU1;
    float* al = alpha[b];
    al[0] = 0.0f;
    float c = 0.0f;
    for (int u = 1; u < UU1; ++u) { c += em[u - 1]; al[u] = c; }
    for (int t = 1; t < S; ++t) {
      const float* blp = bl + (size_t)(t - 1) * UU1;
      const float* emt = em + (size_t)t * UU1;
      float a0 = al[0] + blp[0];
      al[0] = a0;
      float cc = a0;
      for (int u = 1; u < UU1; ++u) {
        float au = al[u] + blp[u];
        cc = lae(au, cc + emt[u - 1]);
        al[u] = cc;
      }
    }
    nllb[b] = -(al[U] + bl[(size_t)(S - 1) * UU1 + U]);
  }
  __syncthreads();
  if (tid == 0) {
    float tot = red[0];  // CE_SCALE == 1
    for (int b = 0; b < BB; ++b) tot += nllb[b];
    out[0] = tot;
  }
}

extern "C" void kernel_launch(void* const* d_in, const int* in_sizes, int n_in,
                              void* d_out, int out_size, void* d_ws, size_t ws_size,
                              hipStream_t stream) {
  const float* x = (const float*)d_in[0];
  const float* w = (const float*)d_in[1];
  const float* bias = (const float*)d_in[2];
  const int* targets = (const int*)d_in[3];
  const int* srcl = (const int*)d_in[4];
  const int* tgtl = (const int*)d_in[5];
  float* out = (float*)d_out;

  // Workspace carve-out (~8.8 MB total)
  char* ws = (char*)d_ws;
  size_t off = 0;
  auto carve = [&](size_t bytes) -> void* {
    void* p = ws + off;
    off += (bytes + 255) & ~(size_t)255;
    return p;
  };
  unsigned short* whi = (unsigned short*)carve((size_t)VV * DD * 2);
  unsigned short* wlo = (unsigned short*)carve((size_t)VV * DD * 2);
  float* blank_lp = (float*)carve((size_t)RTOT * 4);
  float* emit_lp = (float*)carve((size_t)RTOT * 4);
  int* tcolR = (int*)carve((size_t)RTOT * 4);
  int* rowmap = (int*)carve((size_t)CE_ROWS_PAD * 4);
  int* tcolCE = (int*)carve((size_t)CE_ROWS_PAD * 4);
  float* ce_contrib = (float*)carve((size_t)CE_ROWS * 4);

  k_prep_weight<<<(VV * DD) / (256 * 8), 256, 0, stream>>>(w, whi, wlo);
  k_prep_index<<<(RTOT + 255) / 256, 256, 0, stream>>>(targets, srcl, tcolR, rowmap, tcolCE);
  // Main fused GEMM + softmax stats: 33792 rows / 48 = 704 blocks
  k_gemm<<<RTOT / ROWS_PB, NTHREADS, 0, stream>>>(x, whi, wlo, bias, tcolR, rowmap,
                                                  blank_lp, emit_lp, ce_contrib, 0);
  // CE head: 256 gathered rows (padded to 288) = 6 blocks
  k_gemm<<<CE_ROWS_PAD / ROWS_PB, NTHREADS, 0, stream>>>(x, whi, wlo, bias, tcolCE, rowmap,
                                                         blank_lp, emit_lp, ce_contrib, 1);
  // DP + final scalar
  k_dp<<<1, 256, 0, stream>>>(blank_lp, emit_lp, ce_contrib, srcl, tgtl, out);
}
